// MultiStageModel_4_5076651344230
// MI455X (gfx1250) — matrix operand order repaired and checked by reading, not compiled, not run on hardware
//
#include <hip/hip_runtime.h>
#include <hip/hip_bf16.h>
#include <math.h>

// ---------------------------------------------------------------------------
// MI455X / gfx1250 implementation (round 2).
//
// Heavy path: 16 gconv() calls (D_MODEL=64, NODE_N=22, DCT_N=35) over BS=1024,
// fused per-sample into one workgroup with all inter-stage tensors in LDS
// (~311 KB of the WGP's 320 KB), math via v_wmma_f32_16x16x32_f16.
//
// All WMMA fragments are contiguous 16B runs:
//   A-fragment = two 8-half runs at K-offsets {hi*8, 16+hi*8}
//   B-fragment = two 8-half runs at K-offsets {hi*8, 16+hi*8} (B stored [col][K])
//   C-tile     = 8 consecutive M per lane -> single b128 store (K-major dest)
// Global f16 tensors are padded to 40 halves/row (80B, 16B-aligned, zero tail)
// so stage-1 A loads are unconditional global_load_b128.
//
// Input pointer table (setup_inputs dict order: src, then params flattened in
// jax-pytree sorted-key order):
//   0  src (1024*35*66)
//   1  conv70_35.W (35*70)      2  conv70_35.b (35)
//   dec1 @3,  dec2 @19: gcs in order gc1,gc2,gcin,gcout, each {A,T,W,b}
//   gcnb[i] @35+4i: {A(484), T(16), W(9), b(4)}
//   mlps: 51 Wfc 52 alpha 53 beta 54 bfc
//   mlpt: 55 Wfc 56 alpha 57 beta 58 bfc
//   proj1: 59 W 60 b;  proj2: 61 W 62 b
//   63 w1  64 w2  65 w3  66 w4
// ---------------------------------------------------------------------------

typedef _Float16 h16;
typedef __attribute__((ext_vector_type(16))) _Float16 v16h;
typedef __attribute__((ext_vector_type(8)))  _Float16 v8h;
typedef __attribute__((ext_vector_type(8)))  float    v8f;

#define BS_    1024
#define DCTN   35
#define NODEN  22
#define DM     64
#define TPADG  40                 // global f16 row pad (halves): 80B rows
#define VPAD   36                 // padded time axis inside LDS (v dimension)

#define LDS_Y1T ((VPAD*32)*64)    // Y1t[(v*32+n)][c]   : 73728 halves
#define LDS_Y2T ((VPAD*64)*32)    // Y2t[(v*64+d)][m]   : 73728 halves
#define LDS_TLT (48*64)           // Tlt[v][t] (B col-major): 3072 halves
#define LDS_WL  (64*64)           // Wl[d][c] (zero-padded rows d>=outC)
#define LDS_AL  (32*32)           // Al[n][m] zero-padded
#define GCONV_SMEM (int)(((size_t)LDS_Y1T+LDS_Y2T+LDS_TLT+LDS_WL+LDS_AL)*2 + DCTN*4)

#define CAT16(lo, hi_) __builtin_shufflevector((lo), (hi_), \
    0,1,2,3,4,5,6,7,8,9,10,11,12,13,14,15)

__device__ __forceinline__ v8h ld8(const h16* p) { return *(const v8h*)p; }
__device__ __forceinline__ void st8(h16* p, v8h v) { *(v8h*)p = v; }
__device__ __forceinline__ v8h cvt8(v8f a) {
  v8h o;
#pragma unroll
  for (int r = 0; r < 8; ++r) o[r] = (h16)a[r];
  return o;
}
__device__ __forceinline__ v8f wmma16(v16h a, v16h b, v8f c) {
  return __builtin_amdgcn_wmma_f32_16x16x32_f16(false, a, false, b, (short)0,
                                                c, false, false);
}

// ---------------------------------------------------------------------------
// Fused gconv:  out[b,d,n,t] = tanh?( sum_m A[n,m] * sum_c W[c,d] *
//                                     sum_t' X[b,c,m,t'] T[t',t]  + bias[t] )
// One workgroup (512 thr = 16 waves) per batch sample. X optionally += Res.
// outC==64 -> f16 padded output (Hout); outC==3 -> f32 output (Fout).
// ---------------------------------------------------------------------------
__global__ void __launch_bounds__(512)
k_gconv(const h16* __restrict__ Xin, const h16* __restrict__ Res,
        h16* __restrict__ Hout, float* __restrict__ Fout,
        const float* __restrict__ Tg, const float* __restrict__ Wg,
        const float* __restrict__ Ag, const float* __restrict__ bg,
        int outC, int applyTanh)
{
  extern __shared__ char smem[];
  h16*   Y1t = (h16*)smem;             // [(v*32+n)][c=64]
  h16*   Y2t = Y1t + LDS_Y1T;          // [(v*64+d)][m=32]
  h16*   Tlt = Y2t + LDS_Y2T;          // [v=48][t=64]  (B for stage 1)
  h16*   Wl  = Tlt + LDS_TLT;          // [d=64][c=64]  (B for stage 2, per-lane col)
  h16*   Al  = Wl + LDS_WL;            // [n=32][m=32]  (A for stage 3)
  float* bl  = (float*)(Al + LDS_AL);  // [35]

  const int tid  = threadIdx.x;
  const int b    = blockIdx.x;
  const int lane = tid & 31;
  const int wave = tid >> 5;
  const int m16  = lane & 15;
  const int hi   = lane >> 4;
  const v8h zh  = {(h16)0.f,(h16)0.f,(h16)0.f,(h16)0.f,
                   (h16)0.f,(h16)0.f,(h16)0.f,(h16)0.f};
  const v8f z8  = {0.f,0.f,0.f,0.f,0.f,0.f,0.f,0.f};

  // ---- pack parameter matrices (zero-padded) -----------------------------
  for (int i = tid; i < LDS_TLT; i += 512) {       // Tlt[v][t] = T[t][v]
    int v = i >> 6, t = i & 63;
    Tlt[i] = (v < DCTN && t < DCTN) ? (h16)Tg[t * DCTN + v] : (h16)0.f;
  }
  for (int i = tid; i < LDS_WL; i += 512) {        // Wl[d][c] = W[c][d]
    int d = i >> 6, c = i & 63;
    Wl[i] = (d < outC) ? (h16)Wg[c * outC + d] : (h16)0.f;
  }
  for (int i = tid; i < LDS_AL; i += 512) {
    int n = i >> 5, m = i & 31;
    Al[i] = (n < NODEN && m < NODEN) ? (h16)Ag[n * NODEN + m] : (h16)0.f;
  }
  for (int i = tid; i < DCTN; i += 512) bl[i] = bg[i];
  // zero Y1t padding rows (n in [22,32), all v) so no garbage enters stage 2
  for (int i = tid; i < VPAD * 10 * 8; i += 512) {
    int v = i / 80, rem = i % 80;
    int n = 22 + rem / 8, seg = rem % 8;
    st8(Y1t + (size_t)(v * 32 + n) * 64 + seg * 8, zh);
  }
  __syncthreads();

  const size_t xbase = (size_t)b * DM * NODEN * TPADG;

  // ---- stage 1: Y1t[(v,n), c] = sum_t X[(n,c), t] * T[t, v] ---------------
  // rows = n*64+c (1408, 88 tiles, n fixed per tile), v = 3 tiles, K = 2 steps
  for (int tile = wave; tile < 88 * 3; tile += 16) {
    int rt = tile / 3, vt = tile % 3;
    int row = rt * 16 + m16;
    int n = row >> 6, c = row & 63;
    const h16* px = Xin + xbase + (size_t)(c * NODEN + n) * TPADG;
    {   // prefetch the row this wave needs next iteration
      int t2 = tile + 16;
      if (t2 < 88 * 3) {
        int r2 = (t2 / 3) * 16 + m16;
        __builtin_prefetch(Xin + xbase + (size_t)((r2 & 63) * NODEN + (r2 >> 6)) * TPADG, 0, 3);
      }
    }
    v8h a00 = ld8(px + hi * 8);            // K run [hi*8, +8)
    v8h a01 = ld8(px + 16 + hi * 8);       // K run [16+hi*8, +8)
    v8h a10 = ld8(px + 32);                // K run [32, 40): t 32..34 + zero pad
    if (Res) {
      const h16* pr = Res + xbase + (size_t)(c * NODEN + n) * TPADG;
      a00 = a00 + ld8(pr + hi * 8);
      a01 = a01 + ld8(pr + 16 + hi * 8);
      a10 = a10 + ld8(pr + 32);
    }
    if (hi) a10 = zh;                      // K 40..47 is beyond the row -> 0
    const int v = vt * 16 + m16;
    const h16* tb = Tlt + (size_t)v * 64;
    v16h b0 = CAT16(ld8(tb + hi * 8),      ld8(tb + 16 + hi * 8));
    v16h b1 = CAT16(ld8(tb + 32 + hi * 8), ld8(tb + 48 + hi * 8));
    v8f acc = wmma16(CAT16(a00, a01), b0, z8);
    acc     = wmma16(CAT16(a10, zh),  b1, acc);
    if (v < VPAD)   // single contiguous b128 store: 8 consecutive c
      st8(Y1t + (size_t)(v * 32 + n) * 64 + (rt & 3) * 16 + hi * 8, cvt8(acc));
  }
  __syncthreads();

  // ---- stage 2 (transposed): Y2t[(v,d), n] = sum_c Y1t[(v,n), c]*W[c,d] ---
  const int dT = (outC == DM) ? 4 : 1;
  for (int tile = wave; tile < 72 * dT; tile += 16) {
    int rt = tile / dT, dt = tile % dT;                // rows = (v*32+n)
    const h16* ab = Y1t + (size_t)(rt * 16 + m16) * 64;
    const h16* bb = Wl + (size_t)(dt * 16 + m16) * 64; // per-lane column d
    v8f acc = z8;
#pragma unroll
    for (int kk = 0; kk < 2; ++kk) {
      const int kb = kk * 32;
      v16h a  = CAT16(ld8(ab + kb + hi * 8), ld8(ab + kb + 16 + hi * 8));
      v16h bm = CAT16(ld8(bb + kb + hi * 8), ld8(bb + kb + 16 + hi * 8));
      acc = wmma16(a, bm, acc);
    }
    // rows of this tile: v = rt/2 fixed, n = (rt&1)*16 + r + hi*8 consecutive
    st8(Y2t + (size_t)((rt >> 1) * 64 + dt * 16 + m16) * 32 + (rt & 1) * 16 + hi * 8,
        cvt8(acc));
  }
  __syncthreads();

  // ---- stage 3: out[n, (v,d)] = sum_m A[n,m] * Y2t[(v,d), m] + bias -------
  const int dT3 = (outC == DM) ? 4 : 1;
  for (int tile = wave; tile < 2 * VPAD * dT3; tile += 16) {
    int nt = tile / (VPAD * dT3), ct = tile % (VPAD * dT3);
    int v = ct / dT3, dt = ct % dT3;
    const h16* ab = Al + (size_t)(nt * 16 + m16) * 32;
    v16h a  = CAT16(ld8(ab + hi * 8), ld8(ab + 16 + hi * 8));
    const h16* bb = Y2t + (size_t)(v * 64 + dt * 16 + m16) * 32;
    v16h bm = CAT16(ld8(bb + hi * 8), ld8(bb + 16 + hi * 8));
    v8f acc = wmma16(a, bm, z8);
    const int d = dt * 16 + m16;
#pragma unroll
    for (int r = 0; r < 8; ++r) {
      int n = nt * 16 + r + hi * 8;
      if (n >= NODEN) continue;
      if (Fout) {
        if (v < DCTN && d < outC) {
          float val = acc[r] + bl[v];
          if (applyTanh) val = tanhf(val);
          Fout[(((size_t)b * outC + d) * NODEN + n) * DCTN + v] = val;
        }
      } else {
        h16* hb = Hout + (((size_t)b * DM + d) * NODEN + n) * TPADG;
        if (v < DCTN) {
          float val = acc[r] + bl[v];
          if (applyTanh) val = tanhf(val);
          hb[v] = (h16)val;
        } else {           // v == 35: zero the row's pad tail once
          hb[35] = (h16)0.f; hb[36] = (h16)0.f; hb[37] = (h16)0.f;
          hb[38] = (h16)0.f; hb[39] = (h16)0.f;
        }
      }
    }
  }
}

// ---------------------------------------------------------------------------
// Small scalar kernels
// ---------------------------------------------------------------------------
__global__ void k_dct(float* dct) {
  int i = blockIdx.x * blockDim.x + threadIdx.x;
  if (i < DCTN * DCTN) {
    int k = i / DCTN, col = i % DCTN;
    float w = (k == 0) ? sqrtf(1.f / DCTN) : sqrtf(2.f / DCTN);
    dct[i] = w * cosf(3.14159265358979323846f * (col + 0.5f) * k / DCTN);
  }
}

// steps 1-4: gather idx, DCT, reshape, proj1 -> X0[b,c=64,n,TPADG] (f16)
__global__ void k_input(const float* __restrict__ src, const float* __restrict__ dct,
                        const float* __restrict__ Wp, const float* __restrict__ bp,
                        h16* __restrict__ X0) {
  int b = blockIdx.x;
  for (int p = threadIdx.x; p < NODEN * TPADG; p += blockDim.x) {
    int n = p / TPADG, t = p % TPADG;
    if (t >= DCTN) {
      for (int d = 0; d < DM; ++d)
        X0[(((size_t)b * DM + d) * NODEN + n) * TPADG + t] = (h16)0.f;
      continue;
    }
    float xc0 = 0.f, xc1 = 0.f, xc2 = 0.f;
    for (int j = 0; j < DCTN; ++j) {
      int jj = (j < 10) ? j : 9;
      float djt = dct[t * DCTN + j];
      const float* s = src + ((size_t)b * DCTN + jj) * 66 + n * 3;
      xc0 += djt * s[0]; xc1 += djt * s[1]; xc2 += djt * s[2];
    }
    for (int d = 0; d < DM; ++d) {
      float v = bp[d] + xc0 * Wp[d*3] + xc1 * Wp[d*3+1] + xc2 * Wp[d*3+2];
      X0[(((size_t)b * DM + d) * NODEN + n) * TPADG + t] = (h16)v;
    }
  }
}

// proj2: (b,3,22,35) f32 -> (b,64,22,TPADG) f16
__global__ void k_proj(const float* __restrict__ Oin, const float* __restrict__ Wp,
                       const float* __restrict__ bp, h16* __restrict__ X0) {
  int b = blockIdx.x;
  for (int p = threadIdx.x; p < NODEN * TPADG; p += blockDim.x) {
    int n = p / TPADG, t = p % TPADG;
    if (t >= DCTN) {
      for (int d = 0; d < DM; ++d)
        X0[(((size_t)b * DM + d) * NODEN + n) * TPADG + t] = (h16)0.f;
      continue;
    }
    float xc[3];
    for (int c = 0; c < 3; ++c)
      xc[c] = Oin[(((size_t)b * 3 + c) * NODEN + n) * DCTN + t];
    for (int d = 0; d < DM; ++d) {
      float v = bp[d] + xc[0]*Wp[d*3] + xc[1]*Wp[d*3+1] + xc[2]*Wp[d*3+2];
      X0[(((size_t)b * DM + d) * NODEN + n) * TPADG + t] = (h16)v;
    }
  }
}

__global__ void k_accum(const float* __restrict__ Oin, float* __restrict__ COM,
                        const float* w1, const float* w2, const float* w3,
                        const float* w4, int which, int init) {
  size_t idx = (size_t)blockIdx.x * blockDim.x + threadIdx.x;
  const size_t total = (size_t)BS_ * 3 * NODEN * DCTN;
  if (idx < total) {
    float tw = *w1 + *w2 + *w3 + *w4;
    float wr = (which == 0) ? *w1 : (which == 1) ? *w2 : (which == 2) ? *w3 : *w4;
    float v = (wr / tw) * Oin[idx];
    COM[idx] = init ? v : COM[idx] + v;
  }
}

__global__ void k_finhead(const float* __restrict__ COM, float* __restrict__ FIN) {
  size_t idx = (size_t)blockIdx.x * blockDim.x + threadIdx.x;
  const size_t total = (size_t)BS_ * 3 * NODEN * 10;
  if (idx < total) {
    size_t bcn = idx / 10; int t = (int)(idx % 10);
    FIN[bcn * DCTN + t] = COM[bcn * DCTN + t];
  }
}

// shared mlp_block_s + mlp_block_t pipeline; returns last-frame into `frame`
__device__ void mlp_pipeline(const float* inp, const float* diff,
                             const float* Ws, const float* bs_,
                             const float* als, const float* bes,
                             const float* Wt, const float* bt_,
                             const float* alt, const float* bet,
                             float* mid, float* zcol, float* frame) {
  int tid = threadIdx.x, NTH = blockDim.x;
  for (int i = tid; i < 3 * NODEN * 5; i += NTH) {
    int c = i / (NODEN * 5); int m = (i / 5) % NODEN; int t = i % 5;
    float s = bs_[m];
    for (int n = 0; n < NODEN; ++n) s += inp[(c * NODEN + n) * 5 + t] * Ws[n * NODEN + m];
    mid[i] = s;
  }
  __syncthreads();
  for (int i = tid; i < 3 * NODEN; i += NTH) {
    float* y = &mid[i * 5];
    float mean = 0.f; for (int t = 0; t < 5; ++t) mean += y[t]; mean *= 0.2f;
    float var = 0.f;  for (int t = 0; t < 5; ++t) { float d = y[t] - mean; var += d * d; }
    var *= 0.2f;
    float inv = rsqrtf(var + 1e-5f);
    for (int t = 0; t < 5; ++t) {
      float v = (y[t] - mean) * inv * als[t] + bes[t] + inp[i * 5 + t];
      if (diff && t < 4) v += diff[i * 4 + t];
      y[t] = v;
    }
  }
  __syncthreads();
  for (int i = tid; i < 3 * NODEN; i += NTH) {
    float s = bt_[4];
    for (int t = 0; t < 5; ++t) s += mid[i * 5 + t] * Wt[t * 5 + 4];
    zcol[i] = s;
  }
  __syncthreads();
  for (int i = tid; i < 3 * NODEN; i += NTH) {
    int c = i / NODEN, n = i % NODEN;
    float mean = 0.f; for (int k = 0; k < NODEN; ++k) mean += zcol[c * NODEN + k];
    mean /= (float)NODEN;
    float var = 0.f;  for (int k = 0; k < NODEN; ++k) { float d = zcol[c * NODEN + k] - mean; var += d * d; }
    var /= (float)NODEN;
    float inv = rsqrtf(var + 1e-5f);
    frame[i] = mid[i * 5 + 4] + (zcol[i] - mean) * inv * alt[n] + bet[n];
  }
  __syncthreads();
}

// window update for u in 7..11: small gconv on residual + mlp pipeline
__global__ void k_window(const float* __restrict__ COM, const float* __restrict__ src,
                         const float* Tg, const float* Wg, const float* Ag, const float* bg,
                         const float* Ws, const float* bs_, const float* als, const float* bes,
                         const float* Wt, const float* bt_, const float* alt, const float* bet,
                         int u, float* __restrict__ CARRY0, float* __restrict__ FIN) {
  int b = blockIdx.x, cs = u - 4, tid = threadIdx.x;
  __shared__ float inp[3*NODEN*5], mid[3*NODEN*5], diff[3*NODEN*4];
  __shared__ float rs[3*NODEN*4], t1[3*NODEN*4], t2[3*NODEN*4];
  __shared__ float zcol[3*NODEN], frame[3*NODEN];
  for (int i = tid; i < 3 * NODEN * 5; i += blockDim.x) {
    int c = i / (NODEN*5), n = (i / 5) % NODEN, t = i % 5;
    inp[i] = COM[(((size_t)b*3 + c) * NODEN + n) * DCTN + cs + t];
  }
  for (int i = tid; i < 3 * NODEN * 4; i += blockDim.x) {
    int c = i / (NODEN*4), n = (i / 4) % NODEN, t = i % 4;
    int flat = c * (NODEN * DCTN) + n * DCTN + cs + t;    // gt4 = raw reshape
    float g = src[((size_t)b * DCTN + flat / 66) * 66 + flat % 66];
    rs[i] = COM[(((size_t)b*3 + c) * NODEN + n) * DCTN + cs + t] - g;
  }
  __syncthreads();
  for (int i = tid; i < 3 * NODEN * 4; i += blockDim.x) {       // x T
    int cn = i / 4, t = i % 4;
    float s = 0.f; for (int tt = 0; tt < 4; ++tt) s += rs[cn * 4 + tt] * Tg[tt * 4 + t];
    t1[i] = s;
  }
  __syncthreads();
  for (int i = tid; i < 3 * NODEN * 4; i += blockDim.x) {       // x W
    int d = i / (NODEN*4), n = (i / 4) % NODEN, t = i % 4;
    float s = 0.f; for (int c = 0; c < 3; ++c) s += t1[(c * NODEN + n) * 4 + t] * Wg[c * 3 + d];
    t2[i] = s;
  }
  __syncthreads();
  for (int i = tid; i < 3 * NODEN * 4; i += blockDim.x) {       // A x (+bias)
    int c = i / (NODEN*4), n = (i / 4) % NODEN, t = i % 4;
    float s = 0.f; for (int m = 0; m < NODEN; ++m) s += Ag[n * NODEN + m] * t2[(c * NODEN + m) * 4 + t];
    diff[i] = s + bg[t];
  }
  __syncthreads();
  mlp_pipeline(inp, diff, Ws, bs_, als, bes, Wt, bt_, alt, bet, mid, zcol, frame);
  for (int i = tid; i < 3 * NODEN; i += blockDim.x) {
    size_t base = (size_t)b * 3 * NODEN + i;
    CARRY0[base * 4 + (u - 7)] = frame[i];
    if (u == 10) FIN[base * DCTN + 10] = frame[i];
  }
}

__global__ void k_scan(const float* __restrict__ COM, const float* __restrict__ Cin,
                       float* __restrict__ Cout,
                       const float* Ws, const float* bs_, const float* als, const float* bes,
                       const float* Wt, const float* bt_, const float* alt, const float* bet,
                       int step, float* __restrict__ FIN) {
  int b = blockIdx.x, tid = threadIdx.x;
  __shared__ float inp[3*NODEN*5], mid[3*NODEN*5], zcol[3*NODEN], frame[3*NODEN];
  for (int i = tid; i < 3 * NODEN * 5; i += blockDim.x) {
    int cn = i / 5, t = i % 5;
    size_t base = (size_t)b * 3 * NODEN + cn;
    inp[i] = (t < 4) ? Cin[base * 4 + t] : COM[base * DCTN + 11 + step];
  }
  __syncthreads();
  mlp_pipeline(inp, nullptr, Ws, bs_, als, bes, Wt, bt_, alt, bet, mid, zcol, frame);
  for (int i = tid; i < 3 * NODEN; i += blockDim.x) {
    size_t base = (size_t)b * 3 * NODEN + i;
    FIN[base * DCTN + 11 + step] = frame[i];
    for (int j = 0; j < 3; ++j) Cout[base * 4 + j] = Cin[base * 4 + j + 1];
    Cout[base * 4 + 3] = frame[i];
  }
}

// idct (= dct^T, orthonormal), concat with input_gcn1, final 35x70 conv
__global__ void k_final(const float* __restrict__ FIN, const float* __restrict__ src,
                        const float* __restrict__ dct, const float* __restrict__ Wc,
                        const float* __restrict__ bc, float* __restrict__ out) {
  int b = blockIdx.x, tid = threadIdx.x;
  __shared__ float cat[70][66];
  __shared__ float fin2[66][35];
  for (int i = tid; i < 66 * DCTN; i += blockDim.x) {
    int f = i / DCTN, j = i % DCTN, n = f / 3, c = f % 3;
    fin2[f][j] = FIN[(((size_t)b * 3 + c) * NODEN + n) * DCTN + j];
  }
  for (int i = tid; i < DCTN * 66; i += blockDim.x) {
    int s = i / 66, f = i % 66;
    int jj = (s < 10) ? s : 9;
    cat[35 + s][f] = src[((size_t)b * DCTN + jj) * 66 + f];
  }
  __syncthreads();
  for (int i = tid; i < DCTN * 66; i += blockDim.x) {
    int s = i / 66, f = i % 66;
    float acc = 0.f;
    for (int j = 0; j < DCTN; ++j) acc += dct[j * DCTN + s] * fin2[f][j];
    cat[s][f] = acc;
  }
  __syncthreads();
  const size_t rep = (size_t)BS_ * DCTN * 66;
  for (int i = tid; i < DCTN * 66; i += blockDim.x) {
    int o = i / 66, f = i % 66;
    float acc = bc[o];
    for (int s2 = 0; s2 < 70; ++s2) acc += cat[s2][f] * Wc[o * 70 + s2];
    size_t off = ((size_t)b * DCTN + o) * 66 + f;
    for (int r = 0; r < 5; ++r) out[off + (size_t)r * rep] = acc;  // 5-tuple output
  }
}

// ---------------------------------------------------------------------------
extern "C" void kernel_launch(void* const* d_in, const int* in_sizes, int n_in,
                              void* d_out, int out_size, void* d_ws, size_t ws_size,
                              hipStream_t stream) {
  (void)in_sizes; (void)n_in; (void)out_size; (void)ws_size;
  const float* src = (const float*)d_in[0];
  auto P = [&](int i) { return (const float*)d_in[i]; };

  char* ws = (char*)d_ws;
  size_t off = 0;
  auto take = [&](size_t bytes) {
    char* p = ws + off; off += (bytes + 255) & ~(size_t)255; return p;
  };
  const size_t NHP = (size_t)BS_ * DM * NODEN * TPADG;  // padded f16 tensors
  const size_t NC  = (size_t)BS_ * 3 * NODEN * DCTN;    // f32 small tensors
  float* DCT = (float*)take(DCTN * DCTN * sizeof(float));
  h16*   X0  = (h16*)take(NHP * 2);
  h16*   H   = (h16*)take(NHP * 2);
  h16*   Yb  = (h16*)take(NHP * 2);
  float* OUT = (float*)take(NC * 4);
  float* COM = (float*)take(NC * 4);
  float* FIN = (float*)take(NC * 4);
  float* CA  = (float*)take((size_t)BS_ * 3 * NODEN * 4 * 4);
  float* CB  = (float*)take((size_t)BS_ * 3 * NODEN * 4 * 4);

  k_dct<<<dim3((DCTN * DCTN + 255) / 256), dim3(256), 0, stream>>>(DCT);
  k_input<<<dim3(BS_), dim3(256), 0, stream>>>(src, DCT, P(59), P(60), X0);

  for (int r = 0; r < 4; ++r) {
    int g = (r == 0) ? 3 : 19;  // dec1 / dec2; gcs ordered gc1,gc2,gcin,gcout {A,T,W,b}
    const float *A1=P(g+0),*T1=P(g+1),*W1=P(g+2),*b1=P(g+3);
    const float *A2=P(g+4),*T2=P(g+5),*W2=P(g+6),*b2=P(g+7);
    const float *Ai=P(g+8),*Ti=P(g+9),*Wi=P(g+10),*bi=P(g+11);
    const float *Ao=P(g+12),*To=P(g+13),*Wo=P(g+14),*bo=P(g+15);
    // h = tanh(gcin(x))
    k_gconv<<<dim3(BS_), dim3(512), GCONV_SMEM, stream>>>(X0, nullptr, H, nullptr, Ti, Wi, Ai, bi, DM, 1);
    // y = tanh(gc1(h))   (reuses X0 buffer)
    k_gconv<<<dim3(BS_), dim3(512), GCONV_SMEM, stream>>>(H, nullptr, X0, nullptr, T1, W1, A1, b1, DM, 1);
    // y = tanh(gc2(y))
    k_gconv<<<dim3(BS_), dim3(512), GCONV_SMEM, stream>>>(X0, nullptr, Yb, nullptr, T2, W2, A2, b2, DM, 1);
    // out = gcout(h + y)  -> f32
    k_gconv<<<dim3(BS_), dim3(512), GCONV_SMEM, stream>>>(H, Yb, nullptr, OUT, To, Wo, Ao, bo, 3, 0);
    k_accum<<<dim3((unsigned)((NC + 255) / 256)), dim3(256), 0, stream>>>(
        OUT, COM, P(63), P(64), P(65), P(66), r, (r == 0) ? 1 : 0);
    if (r < 3)
      k_proj<<<dim3(BS_), dim3(256), 0, stream>>>(OUT, P(61), P(62), X0);
  }

  k_finhead<<<dim3((unsigned)(((size_t)BS_*3*NODEN*10 + 255) / 256)), dim3(256), 0, stream>>>(COM, FIN);

  for (int i = 0; i < 5; ++i) {
    k_window<<<dim3(BS_), dim3(128), 0, stream>>>(
        COM, src, P(36 + 4*i), P(37 + 4*i), P(35 + 4*i), P(38 + 4*i),
        P(51), P(54), P(52), P(53), P(55), P(58), P(56), P(57),
        7 + i, CA, FIN);
  }
  float* cin = CA; float* cout = CB;
  for (int t = 0; t < 24; ++t) {
    k_scan<<<dim3(BS_), dim3(128), 0, stream>>>(
        COM, cin, cout, P(51), P(54), P(52), P(53), P(55), P(58), P(56), P(57),
        t, FIN);
    float* tmp = cin; cin = cout; cout = tmp;
  }
  k_final<<<dim3(BS_), dim3(256), 0, stream>>>(FIN, src, DCT, P(1), P(2), (float*)d_out);
}
